// SetRepRegressor_20658792693868
// MI455X (gfx1250) — matrix operand order repaired and checked by reading, not compiled, not run on hardware
//
#include <hip/hip_runtime.h>
#include <hip/hip_bf16.h>

typedef __attribute__((ext_vector_type(16))) _Float16 v16h;
typedef __attribute__((ext_vector_type(8)))  float    v8f;

#define NEG_SLOPE 0.01f
#define BN_EPS    1e-5f

// Issue the 8 x 128-bit loads for one 16-row M-tile (this lane's share of the
// 16x64 fp32 A tile, laid out for the 16-bit WMMA A fragment pattern).
__device__ __forceinline__ void load_rows(const float* __restrict__ X,
                                          int b, int t, int m, int hi,
                                          float4* r) {
  const float4* rp4 =
      (const float4*)(X + ((long)b * 256 + (long)t * 16 + m) * 64);
  r[0] = rp4[     2 * hi];  r[1] = rp4[     2 * hi + 1];   // a0 K =  8hi.. +7
  r[2] = rp4[ 4 + 2 * hi];  r[3] = rp4[ 5 + 2 * hi];       // a0 K = 16+8hi..
  r[4] = rp4[ 8 + 2 * hi];  r[5] = rp4[ 9 + 2 * hi];       // a1 K = 32+8hi..
  r[6] = rp4[12 + 2 * hi];  r[7] = rp4[13 + 2 * hi];       // a1 K = 48+8hi..
}

__device__ __forceinline__ void cvt_frags(const float4* r, v16h& a0, v16h& a1) {
#pragma unroll
  for (int i = 0; i < 2; ++i) {
    a0[4 * i + 0]     = (_Float16)r[i].x;     a0[4 * i + 1]     = (_Float16)r[i].y;
    a0[4 * i + 2]     = (_Float16)r[i].z;     a0[4 * i + 3]     = (_Float16)r[i].w;
    a0[4 * i + 8]     = (_Float16)r[2 + i].x; a0[4 * i + 9]     = (_Float16)r[2 + i].y;
    a0[4 * i + 10]    = (_Float16)r[2 + i].z; a0[4 * i + 11]    = (_Float16)r[2 + i].w;
    a1[4 * i + 0]     = (_Float16)r[4 + i].x; a1[4 * i + 1]     = (_Float16)r[4 + i].y;
    a1[4 * i + 2]     = (_Float16)r[4 + i].z; a1[4 * i + 3]     = (_Float16)r[4 + i].w;
    a1[4 * i + 8]     = (_Float16)r[6 + i].x; a1[4 * i + 9]     = (_Float16)r[6 + i].y;
    a1[4 * i + 10]    = (_Float16)r[6 + i].z; a1[4 * i + 11]    = (_Float16)r[6 + i].w;
  }
}

// ---------------------------------------------------------------------------
// Phase 1: fused leaky(X @ Wc) -> max over E=8 -> partial sum over half the
// set, producing P2[512, 64, 2].  GEMM M=131072 N=512 K=64 on
// v_wmma_f32_16x16x32_f16.  One wave per (batch b, h-subblock q, set-half):
// 4096 waves, each holding its 16 B fragments (~128 VGPRs) resident and
// streaming 8 M-tiles with double-buffered prefetch (loads for tile t+1 are
// in flight while tile t converts and runs its 16-WMMA chain).
// leaky_relu is monotone: max_e(leaky(y)) == leaky(max_e(y)).
// No LDS, no atomics, deterministic.
// ---------------------------------------------------------------------------
__global__ __launch_bounds__(256) void srr_phase1(
    const float* __restrict__ X,    // [131072, 64] row-major
    const float* __restrict__ Wc,   // [64, 512]    row-major
    float* __restrict__ P2) {       // [512, 64, 2] partials
  const int tid  = threadIdx.x;
  const int wave = tid >> 5;
  const int lane = tid & 31;
  const int m    = lane & 15;   // row-in-tile (A) / column-in-tile (B,C)
  const int hi   = lane >> 4;   // wave half
  const int g    = blockIdx.x * 8 + wave;  // 0..4095
  const int sh   = g & 1;       // which half of the set (s range)
  const int q    = (g >> 1) & 3;// 16-wide h sub-block of H=64
  const int b    = g >> 3;      // batch index 0..511
  const int t0   = sh * 8;

  // B fragments, loaded once (f32 gather from L2-resident Wc, cvt to f16).
  // 16-bit B layout (32x16): lane holds column N = m; half h holds
  // K = 32*fr + 8*hi + h (h<8), 32*fr + 16 + 8*hi + (h-8) (h>=8).  Column n of tile j = e*4+q is 16*j + m.
  v16h Bf[16];
#pragma unroll
  for (int e = 0; e < 8; ++e) {
    const int n = (e * 4 + q) * 16 + m;
#pragma unroll
    for (int fr = 0; fr < 2; ++fr) {
#pragma unroll
      for (int h = 0; h < 16; ++h) {
        const int K = 32 * fr + (h < 8 ? 8 * hi + h : 16 + 8 * hi + (h - 8));
        Bf[e * 2 + fr][h] = (_Float16)Wc[K * 512 + n];
      }
    }
  }

  float4 rawA[8], rawB[8];
  load_rows(X, b, t0, m, hi, rawA);

  float S = 0.0f;  // running sum over set elements for this lane's column
#pragma unroll
  for (int tt = 0; tt < 8; ++tt) {
    float4* cur = (tt & 1) ? rawB : rawA;
    float4* nxt = (tt & 1) ? rawA : rawB;
    if (tt < 7) load_rows(X, b, t0 + tt + 1, m, hi, nxt);  // prefetch

    v16h a0, a1;
    cvt_frags(cur, a0, a1);

    v8f colmax;
#pragma unroll
    for (int r = 0; r < 8; ++r) colmax[r] = -3.0e38f;

#pragma unroll
    for (int e = 0; e < 8; ++e) {
      v8f acc = {0.f, 0.f, 0.f, 0.f, 0.f, 0.f, 0.f, 0.f};
      acc = __builtin_amdgcn_wmma_f32_16x16x32_f16(
          false, a0, false, Bf[2 * e],     (short)0, acc, false, false);
      acc = __builtin_amdgcn_wmma_f32_16x16x32_f16(
          false, a1, false, Bf[2 * e + 1], (short)0, acc, false, false);
#pragma unroll
      for (int r = 0; r < 8; ++r) colmax[r] = fmaxf(colmax[r], acc[r]);
    }

    // leaky after max (monotone), reduce the 8 rows this half-lane owns
#pragma unroll
    for (int r = 0; r < 8; ++r) {
      float v = colmax[r];
      v = (v >= 0.f) ? v : NEG_SLOPE * v;
      S += v;
    }
  }

  // C layout: lanes 0-15 hold rows M=0..7, lanes 16-31 rows 8..15, same N=m.
  S += __shfl_xor(S, 16, 32);
  if (hi == 0) P2[(b * 64 + q * 16 + m) * 2 + sh] = S;
}

// ---------------------------------------------------------------------------
// Phase 2: sum the two set-half partials, fc1 (64->32), BatchNorm with batch
// stats (biased var) + leaky + fc2 (32->1).  One 512-thread block, one row
// per thread, deterministic LDS tree reductions for mean/var.
// ---------------------------------------------------------------------------
__global__ __launch_bounds__(512) void srr_phase2(
    const float* __restrict__ P2,     // [512, 64, 2]
    const float* __restrict__ fc1_w,  // [32, 64]
    const float* __restrict__ fc1_b,  // [32]
    const float* __restrict__ gamma,  // [32]
    const float* __restrict__ beta,   // [32]
    const float* __restrict__ fc2_w,  // [1, 32]
    const float* __restrict__ fc2_b,  // [1]
    float* __restrict__ out) {        // [512]
  const int r    = threadIdx.x;   // batch row
  const int lane = r & 31;
  const int wave = r >> 5;        // 0..15

  __shared__ float partS[16][32];
  __shared__ float partQ[16][32];
  __shared__ float smu[32];
  __shared__ float srstd[32];

  float pk[64];
#pragma unroll
  for (int k = 0; k < 64; ++k)
    pk[k] = P2[(r * 64 + k) * 2] + P2[(r * 64 + k) * 2 + 1];

  float t1[32];
#pragma unroll
  for (int f = 0; f < 32; ++f) {
    float s = fc1_b[f];
#pragma unroll
    for (int k = 0; k < 64; ++k) s += pk[k] * fc1_w[f * 64 + k];
    t1[f] = s;
  }

  // per-wave reduction of sum / sumsq for each feature
#pragma unroll
  for (int f = 0; f < 32; ++f) {
    float v = t1[f], v2 = v * v;
#pragma unroll
    for (int o = 16; o > 0; o >>= 1) {
      v  += __shfl_xor(v,  o, 32);
      v2 += __shfl_xor(v2, o, 32);
    }
    if (lane == 0) { partS[wave][f] = v; partQ[wave][f] = v2; }
  }
  __syncthreads();

  if (r < 32) {
    float s = 0.f, qq = 0.f;
    for (int w = 0; w < 16; ++w) { s += partS[w][r]; qq += partQ[w][r]; }
    const float mu  = s * (1.0f / 512.0f);
    const float var = qq * (1.0f / 512.0f) - mu * mu;  // biased, matches ref
    smu[r]   = mu;
    srstd[r] = rsqrtf(var + BN_EPS);
  }
  __syncthreads();

  float o = fc2_b[0];
#pragma unroll
  for (int f = 0; f < 32; ++f) {
    float v = (t1[f] - smu[f]) * srstd[f] * gamma[f] + beta[f];
    v = (v >= 0.f) ? v : NEG_SLOPE * v;
    o += v * fc2_w[f];
  }
  out[r] = o;
}

// ---------------------------------------------------------------------------
extern "C" void kernel_launch(void* const* d_in, const int* in_sizes, int n_in,
                              void* d_out, int out_size, void* d_ws,
                              size_t ws_size, hipStream_t stream) {
  (void)in_sizes; (void)n_in; (void)out_size; (void)ws_size;
  const float* X     = (const float*)d_in[0];
  const float* Wc    = (const float*)d_in[1];
  const float* fc1_w = (const float*)d_in[2];
  const float* fc1_b = (const float*)d_in[3];
  const float* gamma = (const float*)d_in[4];
  const float* beta  = (const float*)d_in[5];
  const float* fc2_w = (const float*)d_in[6];
  const float* fc2_b = (const float*)d_in[7];
  float* out = (float*)d_out;
  float* P2  = (float*)d_ws;   // 512*64*2 floats = 256 KB scratch

  // 512 blocks x 8 waves = 4096 waves: one per (batch, h-subblock, set-half).
  srr_phase1<<<512, 256, 0, stream>>>(X, Wc, P2);
  srr_phase2<<<1, 512, 0, stream>>>(P2, fc1_w, fc1_b, gamma, beta,
                                    fc2_w, fc2_b, out);
}
